// lstm_transformer_24034636988661
// MI455X (gfx1250) — matrix operand order repaired and checked by reading, not compiled, not run on hardware
//
#include <hip/hip_runtime.h>
#include <hip/hip_bf16.h>
#include <cstdint>

// ---------------------------------------------------------------------------
// CDNA5 / gfx1250 wave32 WMMA implementation of the transformer+MogLSTM model.
// All GEMMs lower to v_wmma_f32_16x16x32_f16 (f16 in, f32 accumulate).
// Big GEMMs: double-buffered 64x64 block tiles staged with
// GLOBAL_LOAD_ASYNC_TO_LDS_B128 (ASYNCcnt), 4 waves/block, 2x2 WMMA
// fragments per wave (32x32 output per wave).
// ---------------------------------------------------------------------------

typedef _Float16 h16 __attribute__((ext_vector_type(16)));
typedef _Float16 h8  __attribute__((ext_vector_type(8)));
typedef float    f8  __attribute__((ext_vector_type(8)));

struct BMap { long aO, aI, wO, wI, cO, cI; int div; };

__device__ __forceinline__ float wred_sum(float v) {
    for (int m = 16; m >= 1; m >>= 1) v += __shfl_xor(v, m, 32);
    return v;
}
__device__ __forceinline__ float wred_max(float v) {
    for (int m = 16; m >= 1; m >>= 1) v = fmaxf(v, __shfl_xor(v, m, 32));
    return v;
}
__device__ __forceinline__ float sigm(float x) { return 1.0f / (1.0f + expf(-x)); }

__device__ __forceinline__ h16 joinh(h8 lo, h8 hi) {
    return __builtin_shufflevector(lo, hi, 0,1,2,3,4,5,6,7,8,9,10,11,12,13,14,15);
}

// One 16-byte async global->LDS DMA transfer (per-lane addresses), ASYNCcnt.
__device__ __forceinline__ void async_ld16(unsigned ldsAddr, const void* gAddr) {
    asm volatile("global_load_async_to_lds_b128 %0, %1, off"
                 :: "v"(ldsAddr), "v"(gAddr) : "memory");
}
__device__ __forceinline__ void wait_async0() {
    asm volatile("s_wait_asynccnt 0" ::: "memory");
}

// Epilogue helper: accumulator passed BY VALUE so accumulators are never
// address-taken (keeps them in VGPRs across the K loop).
template <bool OUT_HALF, bool RELU, bool ACCUM>
__device__ __forceinline__ void store_tile(f8 a, void* Cv, long offC, int ldc,
                                           int row0, int col, float alpha, float bv)
{
    if (OUT_HALF) {
        _Float16* Ch = (_Float16*)Cv;
        for (int r = 0; r < 8; ++r) {
            float v = a[r] * alpha + bv;
            if (RELU) v = v > 0.0f ? v : 0.0f;
            Ch[offC + (long)(row0 + r) * ldc + col] = (_Float16)v;
        }
    } else {
        float* Cf = (float*)Cv;
        for (int r = 0; r < 8; ++r) {
            long idx = offC + (long)(row0 + r) * ldc + col;
            float v = a[r] * alpha + bv;
            if (ACCUM) v += Cf[idx];
            if (RELU) v = v > 0.0f ? v : 0.0f;
            Cf[idx] = v;
        }
    }
}

// ---------------------------------------------------------------------------
// Direct W-form GEMM (1 wave per 16x16 tile) — used only for K=32 input proj.
// ---------------------------------------------------------------------------
template <bool OUT_HALF, bool RELU, bool ACCUM>
__global__ __launch_bounds__(32)
void k_gemm(const _Float16* __restrict__ A, const _Float16* __restrict__ W,
            const float* __restrict__ bias, void* __restrict__ Cv,
            int K, int lda, int ldw, int ldc, float alpha, BMap bm)
{
    const int lane = threadIdx.x & 31;
    const int g    = lane >> 4;
    const int l    = lane & 15;
    const int tn = blockIdx.x, tm = blockIdx.y, bi = blockIdx.z;

    const long offA = (long)(bi / bm.div) * bm.aO + (long)(bi % bm.div) * bm.aI;
    const long offW = (long)(bi / bm.div) * bm.wO + (long)(bi % bm.div) * bm.wI;
    const long offC = (long)(bi / bm.div) * bm.cO + (long)(bi % bm.div) * bm.cI;

    const _Float16* a0 = A + offA + (long)(tm * 16 + l) * lda + 8 * g;
    const _Float16* w0 = W + offW + (long)(tn * 16 + l) * ldw + 8 * g;

    f8 acc = {};
    for (int k0 = 0; k0 < K; k0 += 32) {
        h8 alo = *(const h8*)(a0 + k0);
        h8 ahi = *(const h8*)(a0 + k0 + 16);
        h8 wlo = *(const h8*)(w0 + k0);
        h8 whi = *(const h8*)(w0 + k0 + 16);
        acc = __builtin_amdgcn_wmma_f32_16x16x32_f16(
            false, joinh(alo, ahi), false, joinh(wlo, whi), (short)0, acc, false, false);
    }

    const int col  = tn * 16 + l;
    const int row0 = tm * 16 + g * 8;
    const float bv = bias ? bias[col] : 0.0f;
    store_tile<OUT_HALF, RELU, ACCUM>(acc, Cv, offC, ldc, row0, col, alpha, bv);
}

// ---------------------------------------------------------------------------
// Async-LDS-staged GEMM: 128 threads = 4 waves, block tile 64(M) x 64(N),
// each wave computes a 32x32 tile as 2x2 WMMA fragments.
// Double-buffered: chunk k+64 streams global->LDS via async DMA (ASYNCcnt)
// while chunk k computes; one workgroup barrier per K-chunk.
// LDS rows padded to 72 halves (144 B) so the 16-lane fragment gathers hit a
// full bank permutation (l*36 mod 64).
// ---------------------------------------------------------------------------
#define SLDA 72
#define LBUF (64 * SLDA)            // halves per buffer
#define LBUF_B (LBUF * 2)           // bytes per buffer

template <bool OUT_HALF, bool RELU, bool ACCUM>
__global__ __launch_bounds__(128)
void k_gemm_lds(const _Float16* __restrict__ A, const _Float16* __restrict__ W,
                const float* __restrict__ bias, void* __restrict__ Cv,
                int K, int lda, int ldw, int ldc, float alpha, BMap bm)
{
    __shared__ _Float16 As[2][LBUF];
    __shared__ _Float16 Bs[2][LBUF];

    const int tid  = threadIdx.x;
    const int lane = tid & 31;
    const int wave = tid >> 5;
    const int wm = wave >> 1, wn = wave & 1;
    const int g = lane >> 4, l = lane & 15;

    const int bi = blockIdx.z;
    const long offA = (long)(bi / bm.div) * bm.aO + (long)(bi % bm.div) * bm.aI;
    const long offW = (long)(bi / bm.div) * bm.wO + (long)(bi % bm.div) * bm.wI;
    const long offC = (long)(bi / bm.div) * bm.cO + (long)(bi % bm.div) * bm.cI;

    const int m0 = blockIdx.y * 64;
    const int n0 = blockIdx.x * 64;

    // cooperative stage-in mapping: 128 threads x 16 B, 16 rows/pass, 4 passes
    const int ldRow   = tid >> 3;        // 0..15
    const int ldChunk = (tid & 7) * 8;   // 0..56 (halves)

    const _Float16* aBase = A + offA + (long)(m0 + ldRow) * lda + ldChunk;
    const _Float16* wBase = W + offW + (long)(n0 + ldRow) * ldw + ldChunk;

    // per-thread LDS destination byte addresses (low 32 bits of flat shared
    // pointer carry the LDS offset)
    const unsigned ldsA0 = (unsigned)(unsigned long long)(uintptr_t)
                               &As[0][ldRow * SLDA + ldChunk];
    const unsigned ldsB0 = (unsigned)(unsigned long long)(uintptr_t)
                               &Bs[0][ldRow * SLDA + ldChunk];
    const unsigned rowStep = 16 * SLDA * 2;   // 16 tile rows in bytes

    f8 acc00 = {}, acc01 = {}, acc10 = {}, acc11 = {};

    // prologue: stage chunk 0 into buffer 0 via async DMA
    #pragma unroll
    for (int p = 0; p < 4; ++p) {
        async_ld16(ldsA0 + p * rowStep, aBase + (long)p * 16 * lda);
        async_ld16(ldsB0 + p * rowStep, wBase + (long)p * 16 * ldw);
    }
    wait_async0();
    __syncthreads();

    int cur = 0;
    for (int k0 = 0; k0 < K; k0 += 64) {
        const bool more = (k0 + 64) < K;
        if (more) {   // stream next chunk into the other buffer while computing
            const unsigned bufOff = (unsigned)(cur ^ 1) * LBUF_B;
            const _Float16* an  = aBase + k0 + 64;
            const _Float16* wn2 = wBase + k0 + 64;
            #pragma unroll
            for (int p = 0; p < 4; ++p) {
                async_ld16(ldsA0 + bufOff + p * rowStep, an  + (long)p * 16 * lda);
                async_ld16(ldsB0 + bufOff + p * rowStep, wn2 + (long)p * 16 * ldw);
            }
        }

        const _Float16* Ac = As[cur];
        const _Float16* Bc = Bs[cur];
        for (int kk = 0; kk < 64; kk += 32) {
            const _Float16* ab0 = &Ac[(wm * 32 + l) * SLDA + kk + 8 * g];
            const _Float16* ab1 = ab0 + 16 * SLDA;
            h16 av0 = joinh(*(const h8*)(ab0), *(const h8*)(ab0 + 16));
            h16 av1 = joinh(*(const h8*)(ab1), *(const h8*)(ab1 + 16));
            const _Float16* wb0 = &Bc[(wn * 32 + l) * SLDA + kk + 8 * g];
            const _Float16* wb1 = wb0 + 16 * SLDA;
            h16 wv0 = joinh(*(const h8*)(wb0), *(const h8*)(wb0 + 16));
            h16 wv1 = joinh(*(const h8*)(wb1), *(const h8*)(wb1 + 16));

            acc00 = __builtin_amdgcn_wmma_f32_16x16x32_f16(false, av0, false, wv0, (short)0, acc00, false, false);
            acc01 = __builtin_amdgcn_wmma_f32_16x16x32_f16(false, av0, false, wv1, (short)0, acc01, false, false);
            acc10 = __builtin_amdgcn_wmma_f32_16x16x32_f16(false, av1, false, wv0, (short)0, acc10, false, false);
            acc11 = __builtin_amdgcn_wmma_f32_16x16x32_f16(false, av1, false, wv1, (short)0, acc11, false, false);
        }

        if (more) {
            wait_async0();      // next buffer fully landed in LDS
            __syncthreads();
        }
        cur ^= 1;
    }

    // epilogue: 4 fragments; D layout: lane -> col (l), rows g*8 + r
    const int colB  = n0 + wn * 32 + l;
    const int rowB  = m0 + wm * 32 + g * 8;
    const float bv0 = bias ? bias[colB]      : 0.0f;
    const float bv1 = bias ? bias[colB + 16] : 0.0f;
    store_tile<OUT_HALF, RELU, ACCUM>(acc00, Cv, offC, ldc, rowB,      colB,      alpha, bv0);
    store_tile<OUT_HALF, RELU, ACCUM>(acc01, Cv, offC, ldc, rowB,      colB + 16, alpha, bv1);
    store_tile<OUT_HALF, RELU, ACCUM>(acc10, Cv, offC, ldc, rowB + 16, colB,      alpha, bv0);
    store_tile<OUT_HALF, RELU, ACCUM>(acc11, Cv, offC, ldc, rowB + 16, colB + 16, alpha, bv1);
}

// ---------------------------------------------------------------------------
// Row kernels (wave per 256-wide row, 8 waves per 256-thread block)
// ---------------------------------------------------------------------------
__global__ __launch_bounds__(256)
void k_add_ln(const float* __restrict__ x, const float* __restrict__ r,
              const float* __restrict__ gg, const float* __restrict__ bb,
              float* __restrict__ yf, _Float16* __restrict__ yh, int rows)
{
    int wave = threadIdx.x >> 5, lane = threadIdx.x & 31;
    long row = (long)blockIdx.x * 8 + wave;
    if (row >= rows) return;
    const float* xr = x + row * 256;
    const float* rr = r + row * 256;
    float v[8]; float s = 0.f, sq = 0.f;
    for (int i = 0; i < 8; ++i) {
        int c = lane * 8 + i;
        float t = xr[c] + rr[c];
        v[i] = t; s += t; sq += t * t;
    }
    s = wred_sum(s); sq = wred_sum(sq);
    float m   = s * (1.0f / 256.0f);
    float var = sq * (1.0f / 256.0f) - m * m;
    float inv = rsqrtf(var + 1e-6f);
    for (int i = 0; i < 8; ++i) {
        int c = lane * 8 + i;
        float o = (v[i] - m) * inv * gg[c] + bb[c];
        yf[row * 256 + c] = o;
        yh[row * 256 + c] = (_Float16)o;
    }
}

__global__ __launch_bounds__(256)
void k_softmax(_Float16* __restrict__ p, long rows)
{
    int wave = threadIdx.x >> 5, lane = threadIdx.x & 31;
    long row = (long)blockIdx.x * 8 + wave;
    if (row >= rows) return;
    _Float16* pr = p + row * 256;
    float v[8], mx = -3.0e38f;
    for (int i = 0; i < 8; ++i) { v[i] = (float)pr[lane * 8 + i]; mx = fmaxf(mx, v[i]); }
    mx = wred_max(mx);
    float s = 0.f;
    for (int i = 0; i < 8; ++i) { v[i] = expf(v[i] - mx); s += v[i]; }
    s = wred_sum(s);
    float inv = 1.0f / s;
    for (int i = 0; i < 8; ++i) pr[lane * 8 + i] = (_Float16)(v[i] * inv);
}

__global__ __launch_bounds__(256)
void k_head(const float* __restrict__ last, const float* __restrict__ w,
            const float* __restrict__ b, float* __restrict__ out, int rows)
{
    int wave = threadIdx.x >> 5, lane = threadIdx.x & 31;
    int row = blockIdx.x * 8 + wave;
    if (row >= rows) return;
    float s = 0.f;
    for (int i = 0; i < 8; ++i) {
        int c = lane * 8 + i;
        s += last[(long)row * 256 + c] * w[c];
    }
    s = wred_sum(s);
    if (lane == 0) out[row] = s + b[0];
}

// ---------------------------------------------------------------------------
// Elementwise / data-movement kernels
// ---------------------------------------------------------------------------
__global__ void k_f2h(const float* __restrict__ in, _Float16* __restrict__ out, int n)
{ int i = blockIdx.x * 256 + threadIdx.x; if (i < n) out[i] = (_Float16)in[i]; }

__global__ void k_zero_f(float* p, int n)
{ int i = blockIdx.x * 256 + threadIdx.x; if (i < n) p[i] = 0.0f; }

__global__ void k_zero_h(_Float16* p, int n)
{ int i = blockIdx.x * 256 + threadIdx.x; if (i < n) p[i] = (_Float16)0.0f; }

// x (B,N,L,C) f32 -> A0 (B*L*N, C) f16, row = (b*L + l)*N + n
__global__ void k_pack_x(const float* __restrict__ x, _Float16* __restrict__ a0, int n)
{
    int i = blockIdx.x * 256 + threadIdx.x; if (i >= n) return;
    int c  = i & 31;  int rr = i >> 5;
    int nn = rr & 255; int bl = rr >> 8;
    int l  = bl & 31;  int b  = bl >> 5;
    long src = (((long)(b * 256 + nn)) * 32 + l) * 32 + c;
    a0[i] = (_Float16)x[src];
}

// encoder (b*L+l, n, e) -> lstm (b*N+n, l, e)
__global__ void k_reshape(const float* __restrict__ src, float* __restrict__ dst, int n)
{
    int i = blockIdx.x * 256 + threadIdx.x; if (i >= n) return;
    int e = i & 255; long row = i >> 8;
    int l = row & 31; long bn = row >> 5;
    int nn = bn & 255; int b = (int)(bn >> 8);
    long s = (((long)(b * 32 + l)) * 256 + nn) * 256 + e;
    dst[i] = src[s];
}

// f32 (R x Cc) -> f16 transposed (Cc x R)
__global__ void k_transpose_w(const float* __restrict__ in, _Float16* __restrict__ out,
                              int R, int Cc)
{
    int i = blockIdx.x * 256 + threadIdx.x; if (i >= R * Cc) return;
    int r = i / Cc, c = i % Cc;
    out[(long)c * R + r] = (_Float16)in[i];
}

// v_h (m*256+n, h*64+d) -> vt (bi=m*4+h: 64 x 256 contiguous blocks)
__global__ void k_transpose_v(const _Float16* __restrict__ v, _Float16* __restrict__ vt, int n)
{
    int i = blockIdx.x * 256 + threadIdx.x; if (i >= n) return;
    int ntok = i & 255; int rem = i >> 8; int d = rem & 63; int bi = rem >> 6;
    int m = bi >> 2, h = bi & 3;
    long src = ((long)(m * 256 + ntok)) * 256 + h * 64 + d;
    vt[i] = v[src];
}

// pull timestep t from seq (rows x 32 x 256) into contiguous f32+f16 copies
__global__ void k_slice_xt(const float* __restrict__ seq, int t,
                           float* __restrict__ xf, _Float16* __restrict__ xh, int n)
{
    int i = blockIdx.x * 256 + threadIdx.x; if (i >= n) return;
    int e = i & 255; long row = i >> 8;
    float v = seq[row * 8192 + (long)t * 256 + e];
    xf[i] = v; xh[i] = (_Float16)v;
}

// mogrifier scaling: t = 2*sigmoid(g)*t  (updates both f32 and f16 copies)
__global__ void k_mog(const float* __restrict__ g, float* __restrict__ tf,
                      _Float16* __restrict__ th, int n)
{
    int i = blockIdx.x * 256 + threadIdx.x; if (i >= n) return;
    float v = 2.0f * sigm(g[i]) * tf[i];
    tf[i] = v; th[i] = (_Float16)v;
}

// LSTM cell pointwise: gates (rows x 1024 = [i|f|c|o]); writes Ct, ht (f32+f16),
// and ht into the output sequence at time t.
__global__ void k_cell(const float* __restrict__ gates, float* __restrict__ Ct,
                       float* __restrict__ hf, _Float16* __restrict__ hh,
                       float* __restrict__ hs, int t, int n)
{
    int i = blockIdx.x * 256 + threadIdx.x; if (i >= n) return;
    int col = i & 255; long row = i >> 8;
    const float* gr = gates + row * 1024;
    float it = sigm(gr[col]);
    float ft = sigm(gr[256 + col]);
    float ct = tanhf(gr[512 + col]);
    float ot = sigm(gr[768 + col]);
    float Cn = ft * Ct[i] + it * ct;
    float hn = ot * tanhf(Cn);
    Ct[i] = Cn; hf[i] = hn; hh[i] = (_Float16)hn;
    hs[row * 8192 + (long)t * 256 + col] = hn;
}

// ---------------------------------------------------------------------------
// Host-side dispatch
// ---------------------------------------------------------------------------
static void launch_gemm(hipStream_t s, const _Float16* A, const _Float16* W,
                        const float* bias, void* C, bool outHalf, bool relu, bool accum,
                        int M, int N, int K, int lda, int ldw, int ldc,
                        float alpha, BMap bm, int batch)
{
    if ((K & 63) == 0 && (M & 63) == 0 && (N & 63) == 0) {
        dim3 grid(N / 64, M / 64, batch), blk(128);
        if (outHalf) {
            if (relu) k_gemm_lds<true,  true,  false><<<grid, blk, 0, s>>>(A, W, bias, C, K, lda, ldw, ldc, alpha, bm);
            else      k_gemm_lds<true,  false, false><<<grid, blk, 0, s>>>(A, W, bias, C, K, lda, ldw, ldc, alpha, bm);
        } else {
            if (accum) k_gemm_lds<false, false, true ><<<grid, blk, 0, s>>>(A, W, bias, C, K, lda, ldw, ldc, alpha, bm);
            else       k_gemm_lds<false, false, false><<<grid, blk, 0, s>>>(A, W, bias, C, K, lda, ldw, ldc, alpha, bm);
        }
    } else {
        dim3 grid(N / 16, M / 16, batch), blk(32);
        if (outHalf) {
            if (relu) k_gemm<true,  true,  false><<<grid, blk, 0, s>>>(A, W, bias, C, K, lda, ldw, ldc, alpha, bm);
            else      k_gemm<true,  false, false><<<grid, blk, 0, s>>>(A, W, bias, C, K, lda, ldw, ldc, alpha, bm);
        } else {
            if (accum) k_gemm<false, false, true ><<<grid, blk, 0, s>>>(A, W, bias, C, K, lda, ldw, ldc, alpha, bm);
            else       k_gemm<false, false, false><<<grid, blk, 0, s>>>(A, W, bias, C, K, lda, ldw, ldc, alpha, bm);
        }
    }
}

extern "C" void kernel_launch(void* const* d_in, const int* in_sizes, int n_in,
                              void* d_out, int out_size, void* d_ws, size_t ws_size,
                              hipStream_t stream)
{
    (void)in_sizes; (void)n_in; (void)out_size; (void)ws_size;

    const int M1 = 65536;   // (B*L) * N  encoder rows
    const int E  = 256, Ff = 1024, H = 256, B2 = 2048;

    // ---- inputs ----
    const float* x     = (const float*)d_in[0];
    const float* lin_w = (const float*)d_in[2];
    const float* lin_b = (const float*)d_in[3];
    const float* mogw  = (const float*)d_in[56];
    const float* mogb  = (const float*)d_in[57];
    const float* outw  = (const float*)d_in[58];
    const float* outb  = (const float*)d_in[59];

    // ---- workspace carve ----
    size_t off = 0;
    char* wsb = (char*)d_ws;
    auto take = [&](size_t bytes) -> void* {
        void* p = (void*)(wsb + off);
        off += (bytes + 255) & ~(size_t)255;
        return p;
    };
    float*    hx_f  = (float*)   take((size_t)M1 * E * 4);   // residual stream
    _Float16* hx_h  = (_Float16*)take((size_t)M1 * E * 2);
    float*    tmp_f = (float*)   take((size_t)M1 * E * 4);
    _Float16* q_h   = (_Float16*)take((size_t)M1 * E * 2);   // + o_h alias / seqA part
    _Float16* k_hb  = (_Float16*)take((size_t)M1 * E * 2);
    _Float16* v_h   = (_Float16*)take((size_t)M1 * E * 2);   // seqB part
    _Float16* vt_h  = (_Float16*)take((size_t)M1 * E * 2);
    _Float16* big_h = (_Float16*)take((size_t)M1 * Ff * 2);  // attn probs / FF hidden
    _Float16* a0_h  = (_Float16*)take((size_t)M1 * 32 * 2);
    // LSTM small buffers
    float*    xt_f    = (float*)take((size_t)B2 * E * 4);
    float*    ht_f    = (float*)take((size_t)B2 * E * 4);
    float*    Ct_f    = (float*)take((size_t)B2 * E * 4);
    float*    mt_f    = (float*)take((size_t)B2 * E * 4);
    float*    gm_f    = (float*)take((size_t)B2 * E * 4);
    float*    gates_f = (float*)take((size_t)B2 * 4 * E * 4);
    float*    last_f  = (float*)take((size_t)B2 * E * 4);
    _Float16* xt_h    = (_Float16*)take((size_t)B2 * E * 2);
    _Float16* ht_h    = (_Float16*)take((size_t)B2 * E * 2);
    _Float16* mt_h    = (_Float16*)take((size_t)B2 * E * 2);
    // half weights
    _Float16* linw_h = (_Float16*)take((size_t)E * 32 * 2);
    _Float16* encw[2][6];
    for (int ly = 0; ly < 2; ++ly) {
        for (int j = 0; j < 4; ++j) encw[ly][j] = (_Float16*)take((size_t)E * E * 2); // wq wk wv wo
        encw[ly][4] = (_Float16*)take((size_t)Ff * E * 2); // w1
        encw[ly][5] = (_Float16*)take((size_t)E * Ff * 2); // w2
    }
    _Float16* lw[2][6];
    for (int ly = 0; ly < 2; ++ly) {
        lw[ly][0] = (_Float16*)take((size_t)4 * H * E * 2); // Wih
        lw[ly][1] = (_Float16*)take((size_t)H * E * 2);     // Wmx
        lw[ly][2] = (_Float16*)take((size_t)H * H * 2);     // Wmh
        lw[ly][3] = (_Float16*)take((size_t)4 * H * H * 2); // Whm
        lw[ly][4] = (_Float16*)take((size_t)E * H * 2);     // QT (E x H)
        lw[ly][5] = (_Float16*)take((size_t)H * E * 2);     // RT (H x E)
    }
    _Float16* mogw_h = (_Float16*)take((size_t)E * H * 2);
    // sequence buffers alias the (now free in LSTM phase) qkv regions
    float* seqA = (float*)q_h;   // spans q_h + k_hb  (64 MB)
    float* seqB = (float*)v_h;   // spans v_h + vt_h  (64 MB)

    const BMap nb{0, 0, 0, 0, 0, 0, 1};
    auto ew = [&](int n) { return dim3((n + 255) / 256); };
    auto f2h = [&](const float* s, _Float16* d, int n) {
        k_f2h<<<ew(n), 256, 0, stream>>>(s, d, n);
    };

    // ================= input projection =================
    k_pack_x<<<ew(M1 * 32), 256, 0, stream>>>(x, a0_h, M1 * 32);
    f2h(lin_w, linw_h, E * 32);
    launch_gemm(stream, a0_h, linw_h, lin_b, hx_f, false, false, false,
                M1, E, 32, 32, 32, E, 1.0f, nb, 1);
    f2h(hx_f, hx_h, M1 * E);

    // ================= encoder layers =================
    for (int ly = 0; ly < 2; ++ly) {
        const int base = 4 + 16 * ly;
        const float* wq = (const float*)d_in[base + 0];
        const float* bq = (const float*)d_in[base + 1];
        const float* wk = (const float*)d_in[base + 2];
        const float* bk = (const float*)d_in[base + 3];
        const float* wv = (const float*)d_in[base + 4];
        const float* bv = (const float*)d_in[base + 5];
        const float* wo = (const float*)d_in[base + 6];
        const float* bo = (const float*)d_in[base + 7];
        const float* w1 = (const float*)d_in[base + 8];
        const float* b1 = (const float*)d_in[base + 9];
        const float* w2 = (const float*)d_in[base + 10];
        const float* b2 = (const float*)d_in[base + 11];
        const float* g1 = (const float*)d_in[base + 12];
        const float* be1 = (const float*)d_in[base + 13];
        const float* g2 = (const float*)d_in[base + 14];
        const float* be2 = (const float*)d_in[base + 15];

        f2h(wq, encw[ly][0], E * E); f2h(wk, encw[ly][1], E * E);
        f2h(wv, encw[ly][2], E * E); f2h(wo, encw[ly][3], E * E);
        f2h(w1, encw[ly][4], Ff * E); f2h(w2, encw[ly][5], E * Ff);

        // Q, K, V projections (f16 outputs feed the attention GEMMs)
        launch_gemm(stream, hx_h, encw[ly][0], bq, q_h,  true, false, false,
                    M1, E, E, E, E, E, 1.0f, nb, 1);
        launch_gemm(stream, hx_h, encw[ly][1], bk, k_hb, true, false, false,
                    M1, E, E, E, E, E, 1.0f, nb, 1);
        launch_gemm(stream, hx_h, encw[ly][2], bv, v_h,  true, false, false,
                    M1, E, E, E, E, E, 1.0f, nb, 1);
        k_transpose_v<<<ew(M1 * E), 256, 0, stream>>>(v_h, vt_h, M1 * E);

        // scores = 1/sqrt(64) * Q @ K^T  (1024 batches of 256x256x64)
        const BMap bmS{65536, 64, 65536, 64, 262144, 65536, 4};
        launch_gemm(stream, q_h, k_hb, nullptr, big_h, true, false, false,
                    256, 256, 64, 256, 256, 256, 0.125f, bmS, 1024);
        k_softmax<<<262144 / 8, 256, 0, stream>>>(big_h, 262144);

        // O = P @ V  (uses pre-transposed V; reuse q_h region for O f16)
        const BMap bmPV{262144, 65536, 65536, 16384, 65536, 64, 4};
        launch_gemm(stream, big_h, vt_h, nullptr, q_h, true, false, false,
                    256, 64, 256, 256, 256, 256, 1.0f, bmPV, 1024);

        // output projection, residual + LN1
        launch_gemm(stream, q_h, encw[ly][3], bo, tmp_f, false, false, false,
                    M1, E, E, E, E, E, 1.0f, nb, 1);
        k_add_ln<<<M1 / 8, 256, 0, stream>>>(hx_f, tmp_f, g1, be1, hx_f, hx_h, M1);

        // FFN: relu(x @ w1^T + b1) @ w2^T + b2; residual + LN2
        launch_gemm(stream, hx_h, encw[ly][4], b1, big_h, true, true, false,
                    M1, Ff, E, E, E, Ff, 1.0f, nb, 1);
        launch_gemm(stream, big_h, encw[ly][5], b2, tmp_f, false, false, false,
                    M1, E, Ff, Ff, Ff, E, 1.0f, nb, 1);
        k_add_ln<<<M1 / 8, 256, 0, stream>>>(hx_f, tmp_f, g2, be2, hx_f, hx_h, M1);
    }

    // ================= reshape to LSTM sequence =================
    k_reshape<<<ew(M1 * E), 256, 0, stream>>>(hx_f, seqA, M1 * E);

    // ================= Mogrifier LSTM layers =================
    for (int ly = 0; ly < 2; ++ly) {
        const int base = 36 + 10 * ly;
        const float* Wih = (const float*)d_in[base + 0];
        const float* bih = (const float*)d_in[base + 1];
        const float* Wmx = (const float*)d_in[base + 2];
        const float* bmx = (const float*)d_in[base + 3];
        const float* Wmh = (const float*)d_in[base + 4];
        const float* bmh = (const float*)d_in[base + 5];
        const float* Whm = (const float*)d_in[base + 6];
        const float* bhm = (const float*)d_in[base + 7];
        const float* Qw  = (const float*)d_in[base + 8];
        const float* Rw  = (const float*)d_in[base + 9];

        f2h(Wih, lw[ly][0], 4 * H * E);
        f2h(Wmx, lw[ly][1], H * E);
        f2h(Wmh, lw[ly][2], H * H);
        f2h(Whm, lw[ly][3], 4 * H * H);
        k_transpose_w<<<ew(H * E), 256, 0, stream>>>(Qw, lw[ly][4], H, E); // Q(HxE)->QT(ExH)
        k_transpose_w<<<ew(E * H), 256, 0, stream>>>(Rw, lw[ly][5], E, H); // R(ExH)->RT(HxE)

        float* sin  = ly ? seqB : seqA;
        float* sout = ly ? seqA : seqB;

        k_zero_f<<<ew(B2 * E), 256, 0, stream>>>(ht_f, B2 * E);
        k_zero_f<<<ew(B2 * E), 256, 0, stream>>>(Ct_f, B2 * E);
        k_zero_h<<<ew(B2 * E), 256, 0, stream>>>(ht_h, B2 * E);

        for (int t = 0; t < 32; ++t) {
            k_slice_xt<<<ew(B2 * E), 256, 0, stream>>>(sin, t, xt_f, xt_h, B2 * E);
            for (int i = 1; i <= 5; ++i) {
                if (i & 1) {  // xt = 2*sigmoid(ht @ Q) * xt   (ht @ QT^T)
                    launch_gemm(stream, ht_h, lw[ly][4], nullptr, gm_f, false, false, false,
                                B2, E, H, H, H, E, 1.0f, nb, 1);
                    k_mog<<<ew(B2 * E), 256, 0, stream>>>(gm_f, xt_f, xt_h, B2 * E);
                } else {      // ht = 2*sigmoid(xt @ R) * ht   (xt @ RT^T)
                    launch_gemm(stream, xt_h, lw[ly][5], nullptr, gm_f, false, false, false,
                                B2, H, E, E, E, H, 1.0f, nb, 1);
                    k_mog<<<ew(B2 * E), 256, 0, stream>>>(gm_f, ht_f, ht_h, B2 * E);
                }
            }
            // mt = xt @ Wmx^T + bmx + ht @ Wmh^T + bmh
            launch_gemm(stream, xt_h, lw[ly][1], bmx, mt_f, false, false, false,
                        B2, H, E, E, E, H, 1.0f, nb, 1);
            launch_gemm(stream, ht_h, lw[ly][2], bmh, mt_f, false, false, true,
                        B2, H, H, H, H, H, 1.0f, nb, 1);
            f2h(mt_f, mt_h, B2 * H);
            // gates = xt @ Wih^T + bih + mt @ Whm^T + bhm
            launch_gemm(stream, xt_h, lw[ly][0], bih, gates_f, false, false, false,
                        B2, 4 * H, E, E, E, 4 * H, 1.0f, nb, 1);
            launch_gemm(stream, mt_h, lw[ly][3], bhm, gates_f, false, false, true,
                        B2, 4 * H, H, H, H, 4 * H, 1.0f, nb, 1);
            k_cell<<<ew(B2 * E), 256, 0, stream>>>(gates_f, Ct_f, ht_f, ht_h, sout, t, B2 * E);
        }
    }

    // ================= output head =================
    // after layer 1, output sequence lives in seqA
    k_slice_xt<<<ew(B2 * E), 256, 0, stream>>>(seqA, 31, xt_f, xt_h, B2 * E);
    f2h(mogw, mogw_h, E * H);
    launch_gemm(stream, xt_h, mogw_h, mogb, last_f, false, false, false,
                B2, E, H, H, H, E, 1.0f, nb, 1);
    k_head<<<B2 / 8, 256, 0, stream>>>(last_f, outw, outb, (float*)d_out, B2);
}
